// MyTransformerDecoder_71313636983118
// MI455X (gfx1250) — matrix operand order repaired and checked by reading, not compiled, not run on hardware
//
#include <hip/hip_runtime.h>
#include <math.h>
#include <stdint.h>

// ---------------- problem constants ----------------
constexpr int kBS   = 1024;
constexpr int kNS   = 21;
constexpr int kH    = 256;
constexpr int kPH   = 25;
constexpr int kENCT = 25;
constexpr int kL    = kBS * kNS;     // 21504 rows
constexpr int kKPAD = 288;           // 16 (inp) + 256 (h) padded to 32-multiple

// ---------------- vector types ----------------
typedef __attribute__((ext_vector_type(16))) __bf16 v16bf;
typedef __attribute__((ext_vector_type(8)))  __bf16 v8bf;
typedef __attribute__((ext_vector_type(8)))  float  v8f;

// ---------------- CDNA5 async copy to LDS ----------------
__device__ __forceinline__ void async_b128(uint32_t lds_off, const void* gaddr)
{
    asm volatile("global_load_async_to_lds_b128 %0, %1, off"
                 :: "v"(lds_off), "v"(gaddr) : "memory");
}
__device__ __forceinline__ void wait_async0()
{
    asm volatile("s_wait_asynccnt 0" ::: "memory");
}

// =====================================================================
// WMMA bf16 GEMM:  C(MxN) = A(MxK,bf16,lda) * Bt(NxK,bf16) + bias, opt relu
// 128x128 tile / block; 8 waves, each wave a 32x64 (2x4 wmma) sub-tile.
// Double-buffered LDS tiles filled with global_load_async_to_lds_b128.
// =====================================================================
__global__ __launch_bounds__(256)
void k_gemm_bf16(const __bf16* __restrict__ A, int lda,
                 const __bf16* __restrict__ Bt,   // N x K (pre-transposed)
                 const float* __restrict__ bias,
                 float* __restrict__ C, int ldc,
                 int M, int N, int K, int relu)
{
    __shared__ __bf16 lA[2][128 * 32];   // [row][k]
    __shared__ __bf16 lB[2][128 * 32];   // [n][k]

    const int tid  = threadIdx.x;
    const int wave = tid >> 5;
    const int lane = tid & 31;
    const int r    = lane & 15;
    const int hi   = lane >> 4;
    const int bm   = blockIdx.x * 128;
    const int bn   = blockIdx.y * 128;
    const int wm   = (wave & 3) * 32;    // wave row offset in tile
    const int wn   = (wave >> 2) * 64;   // wave col offset in tile

    // staging geometry: per wave 16 rows of A and 16 rows of B;
    // one async op = 32 lanes * 16B = 8 rows x 4 quads
    const int lrow = lane >> 2;          // 0..7
    const int lqb  = (lane & 3) * 16;    // byte offset within row

    auto stage = [&](int buf, int k0) {
        #pragma unroll
        for (int op = 0; op < 2; ++op) {
            int row = wave * 16 + op * 8 + lrow;
            const char* ga = (const char*)(A + (size_t)(bm + row) * lda + k0) + lqb;
            uint32_t la = (uint32_t)(uintptr_t)(&lA[buf][row * 32]) + lqb;
            async_b128(la, ga);
            const char* gb = (const char*)(Bt + (size_t)(bn + row) * K + k0) + lqb;
            uint32_t lb = (uint32_t)(uintptr_t)(&lB[buf][row * 32]) + lqb;
            async_b128(lb, gb);
        }
    };

    v8f acc[2][4];
    #pragma unroll
    for (int mi = 0; mi < 2; ++mi)
        #pragma unroll
        for (int ni = 0; ni < 4; ++ni)
            #pragma unroll
            for (int e = 0; e < 8; ++e) acc[mi][ni][e] = 0.f;

    const int nk = K >> 5;
    stage(0, 0);

    for (int ki = 0; ki < nk; ++ki) {
        const int buf = ki & 1;
        wait_async0();           // my async copies into 'buf' are done
        __syncthreads();         // everyone's copies are done
        if (ki + 1 < nk) stage(buf ^ 1, (ki + 1) << 5);

        v16bf afrag[2], bfrag[4];
        #pragma unroll
        for (int mi = 0; mi < 2; ++mi) {
            // A 16x32 layout: lanes 0-15 -> K{0..7,16..23}, lanes 16-31 -> K{8..15,24..31}
            const __bf16* p = &lA[buf][(wm + mi * 16 + r) * 32];
            v8bf lo = *(const v8bf*)(p + 8 * hi);
            v8bf ho = *(const v8bf*)(p + 16 + 8 * hi);
            afrag[mi] = __builtin_shufflevector(lo, ho, 0,1,2,3,4,5,6,7,8,9,10,11,12,13,14,15);
        }
        #pragma unroll
        for (int ni = 0; ni < 4; ++ni) {
            // B 32x16 layout: lane n=l%16; lanes 0-15 -> K{0..7,16..23}, lanes 16-31 -> K{8..15,24..31}
            const __bf16* p = &lB[buf][(wn + ni * 16 + r) * 32];
            v8bf lo = *(const v8bf*)(p + 8 * hi);
            v8bf ho = *(const v8bf*)(p + 16 + 8 * hi);
            bfrag[ni] = __builtin_shufflevector(lo, ho, 0,1,2,3,4,5,6,7,8,9,10,11,12,13,14,15);
        }
        #pragma unroll
        for (int mi = 0; mi < 2; ++mi)
            #pragma unroll
            for (int ni = 0; ni < 4; ++ni)
                acc[mi][ni] = __builtin_amdgcn_wmma_f32_16x16x32_bf16(
                    false, afrag[mi], false, bfrag[ni],
                    (short)0, acc[mi][ni], false, false);
    }

    // epilogue: D layout: col = l%16, row = vgpr + 8*(l/16)
    #pragma unroll
    for (int mi = 0; mi < 2; ++mi)
        #pragma unroll
        for (int ni = 0; ni < 4; ++ni) {
            int col = bn + wn + ni * 16 + r;
            float bv = bias ? bias[col] : 0.f;
            #pragma unroll
            for (int v = 0; v < 8; ++v) {
                int row = bm + wm + mi * 16 + 8 * hi + v;
                float val = acc[mi][ni][v] + bv;
                if (relu) val = fmaxf(val, 0.f);
                if (row < M) C[(size_t)row * ldc + col] = val;
            }
        }
}

// =====================================================================
// Weight packing: fp32 (KxN) -> bf16 transposed (NxK), with concat / pad
// =====================================================================
// [W_ih(16x1024); W_hh(256x1024); pad16] -> dst (1024 x 288)
__global__ void k_pack_ihhhT(const float* __restrict__ Wih, const float* __restrict__ Whh,
                             __bf16* __restrict__ dst)
{
    int i = blockIdx.x * 256 + threadIdx.x;   // over 1024*288
    if (i >= 1024 * kKPAD) return;
    int n = i / kKPAD, k = i % kKPAD;
    float v = (k < 16) ? Wih[(size_t)k * 1024 + n]
            : (k < 272) ? Whh[(size_t)(k - 16) * 1024 + n] : 0.f;
    dst[i] = (__bf16)v;
}

// [A(Kxn1) | B(Kxn2)] -> dst ((n1+n2) x K)
__global__ void k_pack_colsT(const float* __restrict__ A, const float* __restrict__ B,
                             __bf16* __restrict__ dst, int K, int n1, int n2)
{
    int i = blockIdx.x * 256 + threadIdx.x;
    int N = n1 + n2;
    if (i >= N * K) return;
    int n = i / K, k = i % K;
    float v = (n < n1) ? A[(size_t)k * n1 + n] : B[(size_t)k * n2 + (n - n1)];
    dst[i] = (__bf16)v;
}

__global__ void k_cat_bias(const float* __restrict__ a, const float* __restrict__ b,
                           float* __restrict__ dst, int n1, int n2)
{
    int i = blockIdx.x * blockDim.x + threadIdx.x;
    if (i < n1) dst[i] = a[i];
    else if (i < n1 + n2) dst[i] = b[i - n1];
}

__global__ void k_zero_pad(__bf16* __restrict__ mixbuf)
{
    int i = blockIdx.x * 256 + threadIdx.x;   // over L*16 pad cols
    if (i >= kL * 16) return;
    int l = i >> 4, j = i & 15;
    mixbuf[(size_t)l * kKPAD + 272 + j] = (__bf16)0.f;
}

__global__ void k_init_xi(const float* __restrict__ x, float* __restrict__ xi,
                          float* __restrict__ loc_start)
{
    int i = blockIdx.x * 256 + threadIdx.x;   // over L*8 ; x is (BS,1,NS,8)
    if (i >= kL * 8) return;
    int l = i >> 3, d = i & 7;
    float v = x[(size_t)l * 8 + d];
    xi[i] = v;
    if (d < 4) loc_start[(size_t)l * 4 + d] = v;
}

// kh/vh = full_enc(BS,ENC_T,NS,8) -> (s,l,8) @ Wk/Wv + bias
__global__ void k_kv(const float* __restrict__ fenc,
                     const float* __restrict__ Wk, const float* __restrict__ bk,
                     const float* __restrict__ Wv, const float* __restrict__ bv,
                     float* __restrict__ kh, float* __restrict__ vh)
{
    int idx = blockIdx.x * 256 + threadIdx.x;    // over ENC_T * L
    if (idx >= kENCT * kL) return;
    int s = idx / kL, l = idx % kL;
    int b = l / kNS, n = l % kNS;
    const float* e = fenc + (((size_t)b * kENCT + s) * kNS + n) * 8;
    float ev[8];
    #pragma unroll
    for (int j = 0; j < 8; ++j) ev[j] = e[j];
    #pragma unroll
    for (int d = 0; d < 8; ++d) {
        float kk = bk[d], vv = bv[d];
        #pragma unroll
        for (int j = 0; j < 8; ++j) { kk += ev[j] * Wk[j * 8 + d]; vv += ev[j] * Wv[j * 8 + d]; }
        kh[(size_t)idx * 8 + d] = kk;
        vh[(size_t)idx * 8 + d] = vv;
    }
}

// ezmix = G @ concat(z, enc)  -> mixbuf cols [0,64)  (bf16)
__global__ __launch_bounds__(256)
void k_mix_encz(const float* __restrict__ z, const float* __restrict__ enc,
                const float* __restrict__ G, __bf16* __restrict__ mixbuf)
{
    int b = blockIdx.x, tid = threadIdx.x;
    __shared__ float sG[kNS * kNS];
    for (int i = tid; i < kNS * kNS; i += 256) sG[i] = G[i];
    __syncthreads();
    for (int i = tid; i < kNS * 64; i += 256) {
        int n = i >> 6, j = i & 63;
        float s = 0.f;
        for (int m = 0; m < kNS; ++m) {
            float v = (j < 32) ? z[((size_t)b * kNS + m) * 32 + j]
                               : enc[((size_t)b * kNS + m) * 32 + (j - 32)];
            s += sG[n * kNS + m] * v;
        }
        mixbuf[((size_t)b * kNS + n) * kKPAD + j] = (__bf16)s;
    }
}

__global__ void k_split_hc(const float* __restrict__ tmp, float* __restrict__ h,
                           float* __restrict__ c)
{
    size_t i = (size_t)blockIdx.x * 256 + threadIdx.x;   // over L*256
    if (i >= (size_t)kL * 256) return;
    size_t l = i >> 8; int j = (int)(i & 255);
    h[i] = tmp[l * 512 + j];
    c[i] = tmp[l * 512 + 256 + j];
}

// =====================================================================
// Per-step kernels
// =====================================================================
// q = c@W_hd+b ; tiny attention (NH=2, HD=4, S=25) ; inp=[xi, q+o@Wo+bo]
// mixbuf[:,0:16)=Geff@inp ; mixbuf[:,16:272)=Geff@h   (bf16)
__global__ __launch_bounds__(256)
void k_attn_mix(const float* __restrict__ c, const float* __restrict__ h,
                const float* __restrict__ xi,
                const float* __restrict__ kh, const float* __restrict__ vh,
                const float* __restrict__ W_hd, const float* __restrict__ b_hd,
                const float* __restrict__ Wq, const float* __restrict__ bq,
                const float* __restrict__ Wo, const float* __restrict__ bo,
                const float* __restrict__ G, const float* __restrict__ G_add,
                __bf16* __restrict__ mixbuf)
{
    int b = blockIdx.x, tid = threadIdx.x;
    __shared__ float sG[kNS * kNS];
    __shared__ float sH[kNS * kH];
    __shared__ float sq[kNS * 8];
    __shared__ float so[kNS * 8];
    __shared__ float sinp[kNS * 16];

    for (int i = tid; i < kNS * kNS; i += 256) sG[i] = G[i] + G_add[i];
    for (int i = tid; i < kNS * kH; i += 256) sH[i] = h[(size_t)b * kNS * kH + i];
    for (int i = tid; i < kNS * 8; i += 256) {
        int n = i >> 3, d = i & 7;
        const float* crow = c + ((size_t)b * kNS + n) * kH;
        float s = b_hd[d];
        #pragma unroll 8
        for (int k = 0; k < kH; ++k) s += crow[k] * W_hd[k * 8 + d];
        sq[i] = s;
        sinp[n * 16 + d] = xi[((size_t)b * kNS + n) * 8 + d];
    }
    __syncthreads();

    for (int i = tid; i < kNS * 2; i += 256) {   // (n, head)
        int n = i >> 1, hh = i & 1;
        size_t l = (size_t)b * kNS + n;
        float qh[4];
        #pragma unroll
        for (int d = 0; d < 4; ++d) {
            float s = bq[hh * 4 + d];
            #pragma unroll
            for (int e = 0; e < 8; ++e) s += sq[n * 8 + e] * Wq[e * 8 + hh * 4 + d];
            qh[d] = s;
        }
        float sc[kENCT], mx = -3.4e38f;
        for (int t = 0; t < kENCT; ++t) {
            const float* kr = kh + (((size_t)t * kL + l) * 8) + hh * 4;
            float s = (qh[0]*kr[0] + qh[1]*kr[1] + qh[2]*kr[2] + qh[3]*kr[3]) * 0.5f;
            sc[t] = s; mx = fmaxf(mx, s);
        }
        float den = 0.f;
        for (int t = 0; t < kENCT; ++t) { sc[t] = __expf(sc[t] - mx); den += sc[t]; }
        float inv = 1.f / den;
        float o0 = 0, o1 = 0, o2 = 0, o3 = 0;
        for (int t = 0; t < kENCT; ++t) {
            const float* vr = vh + (((size_t)t * kL + l) * 8) + hh * 4;
            float w = sc[t] * inv;
            o0 += w * vr[0]; o1 += w * vr[1]; o2 += w * vr[2]; o3 += w * vr[3];
        }
        so[n * 8 + hh * 4 + 0] = o0; so[n * 8 + hh * 4 + 1] = o1;
        so[n * 8 + hh * 4 + 2] = o2; so[n * 8 + hh * 4 + 3] = o3;
    }
    __syncthreads();

    for (int i = tid; i < kNS * 8; i += 256) {   // att_xi = q + o@Wo + bo
        int n = i >> 3, d = i & 7;
        float s = bo[d];
        #pragma unroll
        for (int e = 0; e < 8; ++e) s += so[n * 8 + e] * Wo[e * 8 + d];
        sinp[n * 16 + 8 + d] = sq[i] + s;
    }
    __syncthreads();

    for (int i = tid; i < kNS * 16; i += 256) {  // inpmix
        int n = i >> 4, j = i & 15;
        float s = 0.f;
        for (int m = 0; m < kNS; ++m) s += sG[n * kNS + m] * sinp[m * 16 + j];
        mixbuf[((size_t)b * kNS + n) * kKPAD + j] = (__bf16)s;
    }
    for (int i = tid; i < kNS * kH; i += 256) {  // hmix
        int n = i >> 8, j = i & 255;
        float s = 0.f;
        for (int m = 0; m < kNS; ++m) s += sG[n * kNS + m] * sH[m * kH + j];
        mixbuf[((size_t)b * kNS + n) * kKPAD + 16 + j] = (__bf16)s;
    }
}

__global__ void k_lstm(const float* __restrict__ gates, float* __restrict__ h,
                       float* __restrict__ c)
{
    size_t i = (size_t)blockIdx.x * 256 + threadIdx.x;   // over L*256
    if (i >= (size_t)kL * kH) return;
    size_t l = i >> 8; int j = (int)(i & 255);
    const float* g = gates + l * 1024;
    float ig = g[j], fg = g[256 + j], gg = g[512 + j], og = g[768 + j];
    float si = 1.f / (1.f + __expf(-ig));
    float sf = 1.f / (1.f + __expf(-fg));
    float so = 1.f / (1.f + __expf(-og));
    float c2 = sf * c[i] + si * tanhf(gg);
    float h2 = so * tanhf(c2);
    c[i] = c2; h[i] = h2;
}

// yimix = G @ relu(h)  -> mixbuf cols [0,256)  (bf16)
__global__ __launch_bounds__(256)
void k_mix_yi(const float* __restrict__ h, const float* __restrict__ G,
              __bf16* __restrict__ mixbuf)
{
    int b = blockIdx.x, tid = threadIdx.x;
    __shared__ float sG[kNS * kNS];
    __shared__ float sY[kNS * kH];
    for (int i = tid; i < kNS * kNS; i += 256) sG[i] = G[i];
    for (int i = tid; i < kNS * kH; i += 256) sY[i] = fmaxf(h[(size_t)b * kNS * kH + i], 0.f);
    __syncthreads();
    for (int i = tid; i < kNS * kH; i += 256) {
        int n = i >> 8, j = i & 255;
        float s = 0.f;
        for (int m = 0; m < kNS; ++m) s += sG[n * kNS + m] * sY[m * kH + j];
        mixbuf[((size_t)b * kNS + n) * kKPAD + j] = (__bf16)s;
    }
}

// head: ymix = G @ yi12 ; loc3/log_z small matmuls ; quaternion ; outputs
__global__ __launch_bounds__(256)
void k_head(const float* __restrict__ yi12, const float* __restrict__ G,
            const float* __restrict__ W_loc, const float* __restrict__ b_loc,
            const float* __restrict__ W_lz, const float* __restrict__ b_lz,
            float* __restrict__ loc_start, float* __restrict__ xi,
            float* __restrict__ out, int t)
{
    int b = blockIdx.x, tid = threadIdx.x;
    __shared__ float sG[kNS * kNS];
    __shared__ float sY[kNS * 128];
    __shared__ float sM[kNS * 128];
    __shared__ float sloc3[kNS * 3];
    __shared__ float slz[kNS * 4];
    for (int i = tid; i < kNS * kNS; i += 256) sG[i] = G[i];
    for (int i = tid; i < kNS * 128; i += 256) sY[i] = yi12[(size_t)b * kNS * 128 + i];
    __syncthreads();
    for (int i = tid; i < kNS * 128; i += 256) {
        int n = i >> 7, j = i & 127;
        float s = 0.f;
        for (int m = 0; m < kNS; ++m) s += sG[n * kNS + m] * sY[m * 128 + j];
        sM[i] = s;
    }
    __syncthreads();
    for (int i = tid; i < kNS * 3; i += 256) {
        int n = i / 3, kk = i % 3;
        float s = b_loc[kk];
        #pragma unroll 8
        for (int e = 0; e < 64; ++e) s += sM[n * 128 + e] * W_loc[e * 3 + kk];
        sloc3[i] = s;
    }
    for (int i = tid; i < kNS * 4; i += 256) {
        int n = i >> 2, kk = i & 3;
        float s = b_lz[kk];
        #pragma unroll 8
        for (int e = 0; e < 64; ++e) s += sM[n * 128 + 64 + e] * W_lz[e * 4 + kk];
        slz[i] = 100.f * s;
    }
    __syncthreads();
    if (tid < kNS) {
        int n = tid;
        size_t l = (size_t)b * kNS + n;
        float qd[4] = {1.f, sloc3[n * 3 + 0], sloc3[n * 3 + 1], sloc3[n * 3 + 2]};
        float inv = rsqrtf(qd[0]*qd[0] + qd[1]*qd[1] + qd[2]*qd[2] + qd[3]*qd[3]);
        #pragma unroll
        for (int d = 0; d < 4; ++d) qd[d] *= inv;
        float qs[4];
        #pragma unroll
        for (int d = 0; d < 4; ++d) qs[d] = loc_start[l * 4 + d];
        float loc[4];
        loc[0] = qs[0]*qd[0] - (qs[1]*qd[1] + qs[2]*qd[2] + qs[3]*qd[3]);
        loc[1] = qs[0]*qd[1] + qd[0]*qs[1] + (qs[2]*qd[3] - qs[3]*qd[2]);
        loc[2] = qs[0]*qd[2] + qd[0]*qs[2] + (qs[3]*qd[1] - qs[1]*qd[3]);
        loc[3] = qs[0]*qd[3] + qd[0]*qs[3] + (qs[1]*qd[2] - qs[2]*qd[1]);
        size_t o1 = (size_t)kBS * kPH * kNS * 4;
        size_t base = (((size_t)b * kPH + t) * kNS + n) * 4;
        #pragma unroll
        for (int d = 0; d < 4; ++d) {
            out[base + d]          = loc[d];
            out[o1 + base + d]     = qd[d];
            out[2 * o1 + base + d] = slz[n * 4 + d];
            loc_start[l * 4 + d]   = loc[d];
            xi[l * 8 + d]          = loc[d];
            xi[l * 8 + 4 + d]      = qd[d];
        }
    }
}

// =====================================================================
// Host orchestration
// =====================================================================
extern "C" void kernel_launch(void* const* d_in, const int* in_sizes, int n_in,
                              void* d_out, int out_size, void* d_ws, size_t ws_size,
                              hipStream_t stream)
{
    (void)in_sizes; (void)n_in; (void)out_size; (void)ws_size;
    const float* x      = (const float*)d_in[0];
    const float* enc    = (const float*)d_in[1];
    const float* fenc   = (const float*)d_in[2];
    const float* z      = (const float*)d_in[3];
    const float* G      = (const float*)d_in[5];
    const float* G_add  = (const float*)d_in[6];
    const float* W_ih   = (const float*)d_in[7];
    const float* W_hh   = (const float*)d_in[8];
    const float* b_lstm = (const float*)d_in[9];
    const float* W_fc   = (const float*)d_in[10];
    const float* b_fc   = (const float*)d_in[11];
    const float* W_fc2  = (const float*)d_in[12];
    const float* b_fc2  = (const float*)d_in[13];
    const float* W_h1   = (const float*)d_in[14];
    const float* b_h1   = (const float*)d_in[15];
    const float* W_h2   = (const float*)d_in[16];
    const float* b_h2   = (const float*)d_in[17];
    const float* W_hd   = (const float*)d_in[18];
    const float* b_hd   = (const float*)d_in[19];
    const float* Wq     = (const float*)d_in[20];
    const float* bq     = (const float*)d_in[21];
    const float* Wk     = (const float*)d_in[22];
    const float* bk     = (const float*)d_in[23];
    const float* Wv     = (const float*)d_in[24];
    const float* bv     = (const float*)d_in[25];
    const float* Wo     = (const float*)d_in[26];
    const float* bo     = (const float*)d_in[27];
    const float* W_loc  = (const float*)d_in[28];
    const float* b_loc  = (const float*)d_in[29];
    const float* W_lz   = (const float*)d_in[30];
    const float* b_lz   = (const float*)d_in[31];
    float* out = (float*)d_out;

    char* p = (char*)d_ws;
    auto bump = [&](size_t bytes) -> void* {
        void* r = (void*)p;
        p += (bytes + 255) & ~((size_t)255);
        return r;
    };
    float*  kh       = (float*)bump((size_t)kENCT * kL * 8 * 4);
    float*  vh       = (float*)bump((size_t)kENCT * kL * 8 * 4);
    float*  hbuf     = (float*)bump((size_t)kL * kH * 4);
    float*  cbuf     = (float*)bump((size_t)kL * kH * 4);
    __bf16* mixbuf   = (__bf16*)bump((size_t)kL * kKPAD * 2);
    float*  gates    = (float*)bump((size_t)kL * 1024 * 4);   // also init h/c tmp (Lx512)
    float*  yi12     = (float*)bump((size_t)kL * 128 * 4);
    float*  xi       = (float*)bump((size_t)kL * 8 * 4);
    float*  locs     = (float*)bump((size_t)kL * 4 * 4);
    __bf16* WbT_ihhh = (__bf16*)bump((size_t)1024 * kKPAD * 2);  // 1024 x 288
    __bf16* WbT_h12  = (__bf16*)bump((size_t)512 * 64 * 2);      // 512 x 64
    __bf16* WbT_fc   = (__bf16*)bump((size_t)128 * 256 * 2);     // 128 x 256
    float*  bias_h12 = (float*)bump(512 * 4);
    float*  bias_fc  = (float*)bump(128 * 4);

    // ---- one-time setup (per launch) ----
    k_pack_ihhhT<<<(1024 * kKPAD + 255) / 256, 256, 0, stream>>>(W_ih, W_hh, WbT_ihhh);
    k_pack_colsT<<<(512 * 64 + 255) / 256, 256, 0, stream>>>(W_h1, W_h2, WbT_h12, 64, 256, 256);
    k_pack_colsT<<<(128 * 256 + 255) / 256, 256, 0, stream>>>(W_fc, W_fc2, WbT_fc, 256, 64, 64);
    k_cat_bias<<<2, 256, 0, stream>>>(b_h1, b_h2, bias_h12, 256, 256);
    k_cat_bias<<<1, 128, 0, stream>>>(b_fc, b_fc2, bias_fc, 64, 64);
    k_zero_pad<<<(kL * 16 + 255) / 256, 256, 0, stream>>>(mixbuf);
    k_init_xi<<<(kL * 8 + 255) / 256, 256, 0, stream>>>(x, xi, locs);
    k_kv<<<(kENCT * kL + 255) / 256, 256, 0, stream>>>(fenc, Wk, bk, Wv, bv, kh, vh);

    // h0/c0: ezmix (Lx64) @ [W_h1|W_h2] (64x512)
    k_mix_encz<<<kBS, 256, 0, stream>>>(z, enc, G, mixbuf);
    k_gemm_bf16<<<dim3(kL / 128, 512 / 128), 256, 0, stream>>>(
        mixbuf, kKPAD, WbT_h12, bias_h12, gates, 512, kL, 512, 64, 0);
    k_split_hc<<<(kL * kH + 255) / 256, 256, 0, stream>>>(gates, hbuf, cbuf);

    // ---- recurrent scan ----
    for (int t = 0; t < kPH; ++t) {
        k_attn_mix<<<kBS, 256, 0, stream>>>(cbuf, hbuf, xi, kh, vh, W_hd, b_hd,
                                            Wq, bq, Wo, bo, G, G_add, mixbuf);
        k_gemm_bf16<<<dim3(kL / 128, 1024 / 128), 256, 0, stream>>>(
            mixbuf, kKPAD, WbT_ihhh, b_lstm, gates, 1024, kL, 1024, kKPAD, 0);
        k_lstm<<<(kL * kH + 255) / 256, 256, 0, stream>>>(gates, hbuf, cbuf);
        k_mix_yi<<<kBS, 256, 0, stream>>>(hbuf, G, mixbuf);
        k_gemm_bf16<<<dim3(kL / 128, 1), 256, 0, stream>>>(
            mixbuf, kKPAD, WbT_fc, bias_fc, yi12, 128, kL, 128, 256, 1);
        k_head<<<kBS, 256, 0, stream>>>(yi12, G, W_loc, b_loc, W_lz, b_lz,
                                        locs, xi, out, t);
    }
}
